// DecoderRNN_58815282152248
// MI455X (gfx1250) — matrix operand order repaired and checked by reading, not compiled, not run on hardware
//
#include <hip/hip_runtime.h>

// Problem constants (from reference)
#define B_ 128
#define H_ 512
#define S_ 64
#define V_ 32000
#define T_ 30
#define G3_ 1536   // 3*H

typedef __attribute__((ext_vector_type(16))) __bf16 v16bf;
typedef __attribute__((ext_vector_type(8)))  __bf16 v8bf;
typedef __attribute__((ext_vector_type(8)))  float  v8f;

__device__ __forceinline__ __bf16 f2bf(float f) { return (__bf16)f; }
__device__ __forceinline__ float sigm(float x) { return 1.f / (1.f + __expf(-x)); }

// ---------------- one-time preprocessing ----------------
__global__ void cvt_f32_bf16_kernel(const float* __restrict__ s, __bf16* __restrict__ d, int n) {
  int i = blockIdx.x * blockDim.x + threadIdx.x;
  int st = gridDim.x * blockDim.x;
  for (; i < n; i += st) d[i] = f2bf(s[i]);
}

__global__ void init_h_kernel(const float* __restrict__ h0, float* __restrict__ h,
                              __bf16* __restrict__ hbf) {
  int i = blockIdx.x * blockDim.x + threadIdx.x;  // 65536 threads total
  float v = h0[i];
  h[i] = v;
  hbf[i] = f2bf(v);
}

// ---------------- embedding + attention (per batch row) ----------------
__global__ void attn_kernel(const float* __restrict__ emb_tab,
                            const int* __restrict__ x0,
                            const unsigned long long* __restrict__ amax,
                            int use_amax,
                            const float* __restrict__ enc,     // [S,B,H]
                            const int* __restrict__ source,    // [S,B]
                            const float* __restrict__ h,       // [B,H] f32
                            __bf16* __restrict__ rnn_in_bf) {  // [B,2H] bf16
  int b = blockIdx.x;
  int t = threadIdx.x;
  __shared__ float s_sc[S_];

  int x = use_amax ? (int)(0xFFFFFFFFu - (unsigned)(amax[b] & 0xFFFFFFFFull)) : x0[b];

  for (int j = t; j < H_; j += 256) {
    float e = emb_tab[(size_t)x * H_ + j];
    rnn_in_bf[b * 2 * H_ + j] = f2bf(e > 0.f ? e : 0.f);
  }

  // scores[s] = enc[s,b,:] . h[b,:]   (4 threads per s)
  int s = t >> 2, part = t & 3;
  const float* eb = enc + ((size_t)s * B_ + b) * H_;
  const float* hb = h + (size_t)b * H_;
  float sum = 0.f;
  int j0 = part * 128;
#pragma unroll 4
  for (int j = j0; j < j0 + 128; ++j) sum += eb[j] * hb[j];
  sum += __shfl_xor(sum, 1);
  sum += __shfl_xor(sum, 2);
  if (part == 0) {
    bool ok = source[s * B_ + b] != 1;  // PAD_IDX == 1
    s_sc[s] = ok ? sum : -1e9f;
  }
  __syncthreads();

  if (t == 0) {  // tiny softmax over 64; 128 blocks run in parallel
    float mx = -3.4e38f;
    for (int i = 0; i < S_; ++i) mx = fmaxf(mx, s_sc[i]);
    float sm = 0.f;
    for (int i = 0; i < S_; ++i) { float e = __expf(s_sc[i] - mx); s_sc[i] = e; sm += e; }
    float inv = 1.f / sm;
    for (int i = 0; i < S_; ++i) s_sc[i] *= inv;
  }
  __syncthreads();

  for (int j = t; j < H_; j += 256) {
    float acc = 0.f;
#pragma unroll 8
    for (int si = 0; si < S_; ++si) acc += s_sc[si] * enc[((size_t)si * B_ + b) * H_ + j];
    rnn_in_bf[b * 2 * H_ + H_ + j] = f2bf(acc);
  }
}

// ---------------- bf16 WMMA GEMM: out[M,N] = A[M,K] @ W[N,K]^T + bias ----------------
// block = 128 threads (4 waves). Block tile: 32(M) x 256(N).
// A slab (32 rows x K) staged to LDS via global_load_async_to_lds_b128 (ASYNCcnt),
// XOR-swizzled so fragment reads spread across all 64 banks.
// Each wave: 2 M-subtiles x 4 N-subtiles -> 8 accumulators; B fragments reused 2x.
// grid = (N/256, M/32). Optional fused greedy-argmax into amax[row].
__global__ void gemm_bf16_kernel(const __bf16* __restrict__ A,
                                 const __bf16* __restrict__ W,
                                 const float* __restrict__ bias,
                                 float* __restrict__ out,
                                 int K, int N,
                                 int do_argmax,
                                 unsigned long long* __restrict__ amax) {
  __shared__ __align__(16) unsigned char sA[32 * 2048];  // 64KB: 32 rows x up to 1024 bf16

  const int lane = threadIdx.x & 31;
  const int wave = threadIdx.x >> 5;
  const int m = lane & 15;
  const int g = lane >> 4;
  const int mbase = blockIdx.y * 32;
  const int nbase = blockIdx.x * 256 + wave * 64;

  const int R  = K >> 3;            // 16B chunks per A row
  const int lr = 31 - __clz(R);     // log2(R)

  // ---- stage A slab (rows mbase..mbase+31, contiguous 64*K bytes) into LDS ----
  {
    const unsigned sbase = (unsigned)(uintptr_t)(&sA[0]);
    const unsigned char* gsrc = (const unsigned char*)(A + (size_t)mbase * K);
    for (int c = threadIdx.x; c < (K << 2); c += 128) {   // 32*R = 4*K chunks
      int row = c >> lr;
      int cc  = c & (R - 1);
      unsigned lds_addr = sbase + (unsigned)(((row << lr) + (cc ^ (row & 15))) << 4);
      const unsigned char* ga = gsrc + ((size_t)c << 4);
      asm volatile("global_load_async_to_lds_b128 %0, %1, off"
                   :: "v"(lds_addr), "v"(ga) : "memory");
    }
    asm volatile("s_wait_asynccnt 0" ::: "memory");
  }
  __syncthreads();

  v8f acc[2][4] = {};
  const __bf16* pw0 = W + (size_t)(nbase + m) * K + g * 16;

  for (int kb = 0; kb < K; kb += 32) {
    // A fragments from LDS (ds_load_b128): lane row = mt*16 + m, chunks kb/8+2g and +1
    const int cb = (kb >> 3) + 2 * g;
    v16bf a[2];
#pragma unroll
    for (int mt = 0; mt < 2; ++mt) {
      int row = mt * 16 + m;
      int r15 = row & 15;
      v8bf a0 = *(const v8bf*)(sA + (((row << lr) + (cb ^ r15)) << 4));
      v8bf a1 = *(const v8bf*)(sA + (((row << lr) + ((cb + 1) ^ r15)) << 4));
      a[mt] = __builtin_shufflevector(a0, a1, 0, 1, 2, 3, 4, 5, 6, 7,
                                      8, 9, 10, 11, 12, 13, 14, 15);
    }
    const __bf16* pb = pw0 + kb;
    __builtin_prefetch(pb + 32, 0, 0);  // global_prefetch_b8: next K chunk of B stream
#pragma unroll
    for (int j = 0; j < 4; ++j) {
      v16bf b = *(const v16bf*)(pb + (size_t)(j * 16) * K);
      acc[0][j] = __builtin_amdgcn_wmma_f32_16x16x32_bf16(false, a[0], false, b, (short)0,
                                                          acc[0][j], false, false);
      acc[1][j] = __builtin_amdgcn_wmma_f32_16x16x32_bf16(false, a[1], false, b, (short)0,
                                                          acc[1][j], false, false);
    }
  }

  // ---- store + optional fused argmax ----
#pragma unroll
  for (int mt = 0; mt < 2; ++mt) {
#pragma unroll
    for (int r = 0; r < 8; ++r) {
      int row = mbase + mt * 16 + g * 8 + r;
      unsigned long long best = 0ull;
#pragma unroll
      for (int j = 0; j < 4; ++j) {
        int col = nbase + j * 16 + m;
        float f = acc[mt][j][r] + bias[col];
        out[(size_t)row * N + col] = f;
        if (do_argmax) {
          unsigned u = __float_as_uint(f);
          unsigned mk = (u & 0x80000000u) ? ~u : (u | 0x80000000u);  // order-preserving
          unsigned long long key = ((unsigned long long)mk << 32) |
                                   (unsigned long long)(0xFFFFFFFFu - (unsigned)col);
          if (key > best) best = key;
        }
      }
      if (do_argmax) {
#pragma unroll
        for (int o = 1; o < 16; o <<= 1) {  // reduce 16 lanes sharing this row
          unsigned lo = (unsigned)best, hi = (unsigned)(best >> 32);
          lo = __shfl_xor(lo, o);
          hi = __shfl_xor(hi, o);
          unsigned long long oth = ((unsigned long long)hi << 32) | lo;
          if (oth > best) best = oth;
        }
        if (m == 0) atomicMax(amax + row, best);
      }
    }
  }
}

// ---------------- GRU gate combine: h' = (1-z)*n + z*h ----------------
__global__ void gru_combine_kernel(const float* __restrict__ gi, const float* __restrict__ gh,
                                   float* __restrict__ h, __bf16* __restrict__ hbf,
                                   unsigned long long* __restrict__ amax) {
  int i = blockIdx.x * blockDim.x + threadIdx.x;  // 65536 = B*H
  if (i < B_) amax[i] = 0ull;                     // reset argmax slots before y-GEMM
  int b = i >> 9, u = i & 511;
  const float* gib = gi + (size_t)b * G3_;
  const float* ghb = gh + (size_t)b * G3_;
  float r = sigm(gib[u] + ghb[u]);
  float z = sigm(gib[512 + u] + ghb[512 + u]);
  float n = tanhf(gib[1024 + u] + r * ghb[1024 + u]);
  float hv = h[i];
  float hn = (1.f - z) * n + z * hv;
  h[i] = hn;
  hbf[i] = f2bf(hn);
}

extern "C" void kernel_launch(void* const* d_in, const int* in_sizes, int n_in,
                              void* d_out, int out_size, void* d_ws, size_t ws_size,
                              hipStream_t stream) {
  const float* h0    = (const float*)d_in[0];
  const int*   x0    = (const int*)d_in[1];
  const float* enc   = (const float*)d_in[2];
  const int*   src   = (const int*)d_in[3];
  const float* emb   = (const float*)d_in[4];
  const float* W_ih  = (const float*)d_in[5];
  const float* W_hh  = (const float*)d_in[6];
  const float* b_ih  = (const float*)d_in[7];
  const float* b_hh  = (const float*)d_in[8];
  const float* W_out = (const float*)d_in[9];
  const float* b_out = (const float*)d_in[10];
  float* out = (float*)d_out;

  char* ws = (char*)d_ws;
  size_t off = 0;
  auto take = [&](size_t bytes) -> char* {
    char* p = ws + off;
    off += (bytes + 255) & ~(size_t)255;
    return p;
  };
  __bf16* Wih_bf  = (__bf16*)take((size_t)G3_ * 1024 * 2);
  __bf16* Whh_bf  = (__bf16*)take((size_t)G3_ * H_ * 2);
  __bf16* Wout_bf = (__bf16*)take((size_t)V_ * H_ * 2);
  __bf16* rnn_bf  = (__bf16*)take((size_t)B_ * 1024 * 2);
  float*  h       = (float*)take((size_t)B_ * H_ * 4);
  __bf16* h_bf    = (__bf16*)take((size_t)B_ * H_ * 2);
  float*  gi      = (float*)take((size_t)B_ * G3_ * 4);
  float*  gh      = (float*)take((size_t)B_ * G3_ * 4);
  unsigned long long* amax = (unsigned long long*)take((size_t)B_ * 8);

  cvt_f32_bf16_kernel<<<2048, 256, 0, stream>>>(W_ih, Wih_bf, G3_ * 1024);
  cvt_f32_bf16_kernel<<<1024, 256, 0, stream>>>(W_hh, Whh_bf, G3_ * H_);
  cvt_f32_bf16_kernel<<<4096, 256, 0, stream>>>(W_out, Wout_bf, V_ * H_);
  init_h_kernel<<<256, 256, 0, stream>>>(h0, h, h_bf);

  for (int t = 0; t < T_; ++t) {
    // 1) embed + attention -> rnn_in (bf16); decodes previous argmax for t>0
    attn_kernel<<<B_, 256, 0, stream>>>(emb, x0, amax, t > 0, enc, src, h, rnn_bf);
    // 2) gi = rnn_in @ W_ih^T + b_ih   [128,1536], K=1024
    gemm_bf16_kernel<<<dim3(6, 4), 128, 0, stream>>>(rnn_bf, Wih_bf, b_ih, gi, 1024, G3_, 0, nullptr);
    // 3) gh = h @ W_hh^T + b_hh        [128,1536], K=512
    gemm_bf16_kernel<<<dim3(6, 4), 128, 0, stream>>>(h_bf, Whh_bf, b_hh, gh, H_, G3_, 0, nullptr);
    // 4) GRU combine -> new h (f32 + bf16); resets amax for this step's y-GEMM
    gru_combine_kernel<<<256, 256, 0, stream>>>(gi, gh, h, h_bf, amax);
    // 5) y = h @ W_out^T + b_out -> d_out[t]; fused greedy argmax into amax
    gemm_bf16_kernel<<<dim3(125, 4), 128, 0, stream>>>(h_bf, Wout_bf, b_out,
                                                       out + (size_t)t * B_ * V_, H_, V_, 1, amax);
  }
}
